// Block_16982300688501
// MI455X (gfx1250) — matrix operand order repaired and checked by reading, not compiled, not run on hardware
//
#include <hip/hip_runtime.h>
#include <hip/hip_bf16.h>
#include <math.h>

typedef _Float16 v16h __attribute__((ext_vector_type(16)));
typedef _Float16 v8h  __attribute__((ext_vector_type(8)));
typedef float    v8f  __attribute__((ext_vector_type(8)));

static __device__ __forceinline__ v8f wmma16(v16h a, v16h b, v8f c) {
    return __builtin_amdgcn_wmma_f32_16x16x32_f16(false, a, false, b, (short)0, c, false, false);
}

// ---------------------------------------------------------------------------
// A fragment: 16x32 f16, row-major source, leading dim lda (halves).
// Lane l: m=l&15, hi=l>>4. halves[0..15]=A[m][k0+hi*16..]
// ---------------------------------------------------------------------------
static __device__ __forceinline__ v16h load_a_frag(const _Float16* __restrict__ A,
                                                   size_t lda, int m0, int k0, int lane) {
    int m  = lane & 15;
    int hi = lane >> 4;
    const _Float16* p = A + (size_t)(m0 + m) * lda + k0 + hi * 16;
    v8h lo = *(const v8h*)(p);
    v8h hh = *(const v8h*)(p + 8);
    v16h r;
#pragma unroll
    for (int i = 0; i < 8; ++i) { r[i] = lo[i]; r[8 + i] = hh[i]; }
    return r;
}

// ---------------------------------------------------------------------------
// B fragment: 32x16 = (W[n0..n0+15][k0..k0+31])^T, W row-major [N,K].
// Lane l: n=l&15, koff=(l>=16)*16. halves[t]=W[n][k0+koff+t] (one 32B load).
// ---------------------------------------------------------------------------
static __device__ __forceinline__ v16h load_b_frag(const _Float16* __restrict__ W,
                                                   size_t ldw, int n0, int k0, int lane) {
    int n    = lane & 15;
    int koff = (lane >> 4) * 16;
    return *(const v16h*)(W + (size_t)(n0 + n) * ldw + k0 + koff);
}

// ---------------------------------------------------------------------------
__global__ void convert_kernel(const float* __restrict__ in, _Float16* __restrict__ out, int n) {
    int i = blockIdx.x * blockDim.x + threadIdx.x;
    if (i < n) out[i] = (_Float16)in[i];
}

// ---------------------------------------------------------------------------
// LayerNorm: one 256-thread block per row of 1024, f32 in -> f16 out
// ---------------------------------------------------------------------------
__global__ void ln_kernel(const float* __restrict__ x, const float* __restrict__ gw,
                          const float* __restrict__ bw, _Float16* __restrict__ out) {
    const int Dn = 1024;
    int row = blockIdx.x;
    const float* xr = x + (size_t)row * Dn;
    float v[4];
    float s = 0.f, sq = 0.f;
#pragma unroll
    for (int i = 0; i < 4; ++i) {
        v[i] = xr[threadIdx.x + 256 * i];
        s  += v[i];
        sq += v[i] * v[i];
    }
#pragma unroll
    for (int o = 16; o > 0; o >>= 1) {
        s  += __shfl_down(s, o, 32);
        sq += __shfl_down(sq, o, 32);
    }
    __shared__ float ss[8], ssq[8], stat[2];
    int wid = threadIdx.x >> 5;
    if ((threadIdx.x & 31) == 0) { ss[wid] = s; ssq[wid] = sq; }
    __syncthreads();
    if (threadIdx.x == 0) {
        float ts = 0.f, tq = 0.f;
        for (int i = 0; i < 8; ++i) { ts += ss[i]; tq += ssq[i]; }
        float mu  = ts / Dn;
        float var = tq / Dn - mu * mu;
        stat[0] = mu;
        stat[1] = rsqrtf(var + 1e-6f);
    }
    __syncthreads();
    float mu = stat[0], rstd = stat[1];
    _Float16* orow = out + (size_t)row * Dn;
#pragma unroll
    for (int i = 0; i < 4; ++i) {
        int idx = threadIdx.x + 256 * i;
        orow[idx] = (_Float16)((v[i] - mu) * rstd * gw[idx] + bw[idx]);
    }
}

// ---------------------------------------------------------------------------
// GEMM: out[M,N] = A[M,K] @ W[N,K]^T + bias (+gelu) (+resid).
// Block = (32,4): 4 waves, each a 16x64 tile (A-frag reused 4x). Grid (N/256, M/16).
// ---------------------------------------------------------------------------
template <int K, bool OUT16, bool GELU, bool RESID>
__global__ void gemm_kernel(const _Float16* __restrict__ A, const _Float16* __restrict__ W,
                            const float* __restrict__ bias, const float* __restrict__ resid,
                            _Float16* __restrict__ out16, float* __restrict__ out32,
                            int M, int N) {
    int lane = threadIdx.x;
    int n0 = blockIdx.x * 256 + threadIdx.y * 64;
    int m0 = blockIdx.y * 16;
    const _Float16* arow = A + (size_t)(m0 + (lane & 15)) * K;
    const _Float16* wrow = W + (size_t)(n0 + (lane & 15)) * K;
    v8f acc[4] = {};
#pragma unroll 2
    for (int k0 = 0; k0 < K; k0 += 32) {
        if (k0 + 64 < K) {   // prefetch 2 k-tiles ahead -> global_prefetch_b8
            __builtin_prefetch(arow + k0 + 64, 0, 1);
            __builtin_prefetch(wrow + k0 + 64, 0, 1);
        }
        v16h a = load_a_frag(A, (size_t)K, m0, k0, lane);
#pragma unroll
        for (int c = 0; c < 4; ++c) {
            v16h b = load_b_frag(W, (size_t)K, n0 + c * 16, k0, lane);
            acc[c] = wmma16(a, b, acc[c]);
        }
    }
    int n = lane & 15;
    int g = lane >> 4;
#pragma unroll
    for (int c = 0; c < 4; ++c) {
        int col = n0 + c * 16 + n;
        float bv = bias[col];
#pragma unroll
        for (int r = 0; r < 8; ++r) {
            int row = m0 + r + 8 * g;
            float v = acc[c][r] + bv;
            if (GELU) v = 0.5f * v * (1.0f + erff(v * 0.70710678118f));
            if (RESID) v += resid[(size_t)row * N + col];
            if (OUT16) out16[(size_t)row * N + col] = (_Float16)v;
            else       out32[(size_t)row * N + col] = v;
        }
    }
}

// ---------------------------------------------------------------------------
// Flash attention, transposed-score formulation.
// qkv: f16 [B*S, 3072], feature = which*1024 + h*64 + hd.  ctx: f16 [B*S, 1024].
// One wave = 16 query rows of one (b,h). Block=(32,4) -> 4 tiles per block.
//
// S^T tile = K_block(16x64) @ Q^T(64x16): C layout puts query = lane&15 and
// 16 keys (8g+r, 16+8g+r) in each lane's registers -> softmax needs only ONE
// cross-lane shuffle, and exp(P) lands exactly in the A-fragment layout for
// the ctx WMMA. V is staged TRANSPOSED in LDS (scatter b16 stores at write
// time) so every V B-fragment is two contiguous 16B LDS loads.
// ---------------------------------------------------------------------------
__global__ void attn_kernel(const _Float16* __restrict__ qkv, _Float16* __restrict__ ctx) {
    const int S = 1024;
    const size_t D3 = 3072;
    int lane = threadIdx.x;
    int wv = threadIdx.y;
    int bid = blockIdx.x * 4 + wv;           // b*H*(S/16) + ...
    int mt = bid & 63;
    int h  = (bid >> 6) & 15;
    int b  = bid >> 10;
    int n = lane & 15;
    int g = lane >> 4;

    __shared__ _Float16 lds_vt[4][64 * 32];  // [hd][key], per wave
    __shared__ float    lds_stat[4][16];

    const _Float16* qp  = qkv + (size_t)(b * S + mt * 16) * D3 + h * 64;         // Q
    const _Float16* kbp = qkv + (size_t)(b * S) * D3 + 1024 + h * 64;            // K
    const _Float16* vbp = qkv + (size_t)(b * S) * D3 + 2048 + h * 64;            // V

    // Q as B fragments (loop invariant): B[k][n] = Q[query n][k]
    v16h qf0, qf1;
    {
        const _Float16* p = qp + (size_t)n * D3;
        qf0 = *(const v16h*)(p + g * 16);          // hd 0..31 chunk
        qf1 = *(const v16h*)(p + 32 + g * 16);     // hd 32..63 chunk
    }

    float m_run = -__builtin_inff();
    float l_run = 0.f;
    v8f acc[4] = {};

    for (int j = 0; j < S / 32; ++j) {
        int key0 = j * 32;
        // stage V block transposed: lane owns key row `lane`, scatters 64 hd
        // values into lds_vt[hd][lane] (immediate-offset b16 stores)
        {
            const _Float16* vp = vbp + (size_t)(key0 + lane) * D3;
            v16h va = *(const v16h*)(vp);
            v16h vb = *(const v16h*)(vp + 16);
            v16h vc = *(const v16h*)(vp + 32);
            v16h vd = *(const v16h*)(vp + 48);
#pragma unroll
            for (int i = 0; i < 16; ++i) {
                lds_vt[wv][(i)      * 32 + lane] = va[i];
                lds_vt[wv][(16 + i) * 32 + lane] = vb[i];
                lds_vt[wv][(32 + i) * 32 + lane] = vc[i];
                lds_vt[wv][(48 + i) * 32 + lane] = vd[i];
            }
        }
        // S^T tiles: A = K rows (m = key in half-tile), B = Q^T
        v8f sf[2];
#pragma unroll
        for (int hN = 0; hN < 2; ++hN) {
            const _Float16* kr = kbp + (size_t)(key0 + hN * 16 + n) * D3;
            v8h a0l = *(const v8h*)(kr + g * 16);
            v8h a0h = *(const v8h*)(kr + 8 + g * 16);
            v8h a1l = *(const v8h*)(kr + 32 + g * 16);
            v8h a1h = *(const v8h*)(kr + 40 + g * 16);
            v16h ka0, ka1;
#pragma unroll
            for (int i = 0; i < 8; ++i) {
                ka0[i] = a0l[i]; ka0[8 + i] = a0h[i];
                ka1[i] = a1l[i]; ka1[8 + i] = a1h[i];
            }
            v8f c = {};
            c = wmma16(ka0, qf0, c);
            c = wmma16(ka1, qf1, c);
            sf[hN] = c;
        }
        // online softmax: lane = query n; holds keys 8g+r (tile0), 16+8g+r (tile1)
        float mx = -__builtin_inff();
#pragma unroll
        for (int r = 0; r < 8; ++r) {
            float s0 = sf[0][r] * 0.125f;
            float s1 = sf[1][r] * 0.125f;
            sf[0][r] = s0; sf[1][r] = s1;
            mx = fmaxf(mx, fmaxf(s0, s1));
        }
        mx = fmaxf(mx, __shfl_xor(mx, 16, 32));    // combine with partner half
        float nm   = fmaxf(m_run, mx);
        float corr = __expf(m_run - nm);
        float rs = 0.f;
        v16h pf;                                   // P in A-fragment layout
#pragma unroll
        for (int r = 0; r < 8; ++r) {
            float p0 = __expf(sf[0][r] - nm);
            float p1 = __expf(sf[1][r] - nm);
            rs += p0 + p1;
            pf[r]     = (_Float16)p0;              // key 8g+r
            pf[8 + r] = (_Float16)p1;              // key 16+8g+r
        }
        rs += __shfl_xor(rs, 16, 32);
        l_run = l_run * corr + rs;
        m_run = nm;
        // broadcast corr (per query) to C-layout lanes via tiny LDS array
        if (lane < 16) lds_stat[wv][lane] = corr;
        __syncthreads();
        float cc[8];
#pragma unroll
        for (int r = 0; r < 8; ++r) cc[r] = lds_stat[wv][8 * g + r];
#pragma unroll
        for (int c2 = 0; c2 < 4; ++c2)
#pragma unroll
            for (int r = 0; r < 8; ++r) acc[c2][r] *= cc[r];
        // ctx += P @ V : B-frag = keys 8g..8g+7 and 16+8g..16+8g+7 of lds_vt row
        int koff = g * 8;
#pragma unroll
        for (int c2 = 0; c2 < 4; ++c2) {
            v8h vl = *(const v8h*)(&lds_vt[wv][(c2 * 16 + n) * 32 + koff]);
            v8h vh = *(const v8h*)(&lds_vt[wv][(c2 * 16 + n) * 32 + 16 + koff]);
            v16h vf;
#pragma unroll
            for (int i = 0; i < 8; ++i) { vf[i] = vl[i]; vf[8 + i] = vh[i]; }
            acc[c2] = wmma16(pf, vf, acc[c2]);
        }
        __syncthreads();
    }

    // normalize: broadcast 1/l per query to C-layout lanes, then write ctx
    if (lane < 16) lds_stat[wv][lane] = 1.0f / l_run;
    __syncthreads();
    float inv[8];
#pragma unroll
    for (int r = 0; r < 8; ++r) inv[r] = lds_stat[wv][8 * g + r];
    _Float16* crow = ctx + (size_t)(b * S + mt * 16) * 1024 + h * 64;
#pragma unroll
    for (int r = 0; r < 8; ++r) {
        int row = r + 8 * g;
#pragma unroll
        for (int c2 = 0; c2 < 4; ++c2)
            crow[(size_t)row * 1024 + c2 * 16 + n] = (_Float16)(acc[c2][r] * inv[r]);
    }
}

// ---------------------------------------------------------------------------
extern "C" void kernel_launch(void* const* d_in, const int* in_sizes, int n_in,
                              void* d_out, int out_size, void* d_ws, size_t ws_size,
                              hipStream_t stream) {
    (void)in_sizes; (void)n_in; (void)out_size; (void)ws_size;
    const float* x      = (const float*)d_in[0];
    const float* qkv_w  = (const float*)d_in[1];
    const float* qkv_b  = (const float*)d_in[2];
    const float* proj_w = (const float*)d_in[3];
    const float* proj_b = (const float*)d_in[4];
    const float* fc1_w  = (const float*)d_in[5];
    const float* fc1_b  = (const float*)d_in[6];
    const float* fc2_w  = (const float*)d_in[7];
    const float* fc2_b  = (const float*)d_in[8];
    const float* ln1_g  = (const float*)d_in[9];
    const float* ln1_b  = (const float*)d_in[10];
    const float* ln2_g  = (const float*)d_in[11];
    const float* ln2_b  = (const float*)d_in[12];
    float* out = (float*)d_out;

    const int M = 8 * 1024;   // B*S
    char* ws = (char*)d_ws;
    size_t off = 0;
    auto alloc = [&](size_t bytes) -> char* {
        char* p = ws + off;
        off += (bytes + 255) & ~(size_t)255;
        return p;
    };
    _Float16* wqkv16  = (_Float16*)alloc((size_t)3072 * 1024 * 2);
    _Float16* wproj16 = (_Float16*)alloc((size_t)1024 * 1024 * 2);
    _Float16* wfc1    = (_Float16*)alloc((size_t)4096 * 1024 * 2);
    _Float16* wfc2    = (_Float16*)alloc((size_t)1024 * 4096 * 2);
    _Float16* h1      = (_Float16*)alloc((size_t)M * 1024 * 2);
    _Float16* qkvbuf  = (_Float16*)alloc((size_t)M * 3072 * 2);
    _Float16* ctxbuf  = (_Float16*)alloc((size_t)M * 1024 * 2);
    float*    x2      = (float*)   alloc((size_t)M * 1024 * 4);
    _Float16* h2      = (_Float16*)alloc((size_t)M * 1024 * 2);
    _Float16* fc1o    = (_Float16*)alloc((size_t)M * 4096 * 2);

    // weight conversions
    convert_kernel<<<(3072 * 1024 + 255) / 256, 256, 0, stream>>>(qkv_w, wqkv16, 3072 * 1024);
    convert_kernel<<<(1024 * 1024 + 255) / 256, 256, 0, stream>>>(proj_w, wproj16, 1024 * 1024);
    convert_kernel<<<(4096 * 1024 + 255) / 256, 256, 0, stream>>>(fc1_w, wfc1, 4096 * 1024);
    convert_kernel<<<(4096 * 1024 + 255) / 256, 256, 0, stream>>>(fc2_w, wfc2, 4096 * 1024);

    dim3 gblk(32, 4);

    // LN1 -> h1
    ln_kernel<<<M, 256, 0, stream>>>(x, ln1_g, ln1_b, h1);
    // QKV projection: [M,3072]
    gemm_kernel<1024, true, false, false><<<dim3(3072 / 256, M / 16), gblk, 0, stream>>>(
        h1, wqkv16, qkv_b, nullptr, qkvbuf, nullptr, M, 3072);
    // flash attention -> ctx
    attn_kernel<<<8 * 16 * 64 / 4, gblk, 0, stream>>>(qkvbuf, ctxbuf);
    // proj + residual -> x2 (fp32)
    gemm_kernel<1024, false, false, true><<<dim3(1024 / 256, M / 16), gblk, 0, stream>>>(
        ctxbuf, wproj16, proj_b, x, nullptr, x2, M, 1024);
    // LN2 -> h2
    ln_kernel<<<M, 256, 0, stream>>>(x2, ln2_g, ln2_b, h2);
    // fc1 + exact GELU -> f16
    gemm_kernel<1024, true, true, false><<<dim3(4096 / 256, M / 16), gblk, 0, stream>>>(
        h2, wfc1, fc1_b, nullptr, fc1o, nullptr, M, 4096);
    // fc2 + residual -> out (fp32)
    gemm_kernel<4096, false, false, true><<<dim3(1024 / 256, M / 16), gblk, 0, stream>>>(
        fc1o, wfc2, fc2_b, x2, nullptr, out, M, 1024);
}
